// IdiomExtractor_62672162783559
// MI455X (gfx1250) — matrix operand order repaired and checked by reading, not compiled, not run on hardware
//
#include <hip/hip_runtime.h>

#define DEV static __device__ __forceinline__

typedef unsigned short u16;
typedef __attribute__((ext_vector_type(8)))  __bf16 v8bf;
typedef __attribute__((ext_vector_type(16))) __bf16 v16bf;
typedef __attribute__((ext_vector_type(8)))  float  v8f;

// ---------------- problem constants ----------------
#define BB 64
#define SS 512
#define DD 768
#define HH 384
#define G4H 1536          // 4*H
#define BS 32768          // B*S
#define CC 3

// ---------------- helpers ----------------
DEV v8bf ld8(const u16* p) { return *reinterpret_cast<const v8bf*>(p); }
DEV v16bf mk16(v8bf lo, v8bf hi) {
  return __builtin_shufflevector(lo, hi, 0,1,2,3,4,5,6,7,8,9,10,11,12,13,14,15);
}
DEV v8f wmma_bf16(v16bf a, v16bf b, v8f c) {
  return __builtin_amdgcn_wmma_f32_16x16x32_bf16(false, a, false, b, (short)0, c, false, false);
}
DEV u16 f2bf(float f) {
  unsigned u = __float_as_uint(f);
  u += 0x7fffu + ((u >> 16) & 1u);
  return (u16)(u >> 16);
}
DEV float bf2f(u16 h) { return __uint_as_float(((unsigned)h) << 16); }
DEV float sigf(float x) { return 1.f / (1.f + __expf(-x)); }
DEV float lse3(float a, float b, float c) {
  float m = fmaxf(a, fmaxf(b, c));
  return m + __logf(__expf(a - m) + __expf(b - m) + __expf(c - m));
}

// A-fragment (16x32 bf16, M rows = batch rows): chunks at k+half*8 and +16
DEV v16bf loadA(const u16* base, int row, int ld, int k, int lane) {
  int half = lane >> 4;
  const u16* p = base + row * ld + k + half * 8;
  return mk16(ld8(p), ld8(p + 16));
}
// B-fragment (32x16 bf16, B = W^T, W row-major N x K): chunks at k+half*8 and +16
DEV v16bf loadB(const u16* Wrow, int k, int lane) {
  int half = lane >> 4;
  const u16* p = Wrow + k + half * 8;
  return mk16(ld8(p), ld8(p + 16));
}

// ---------------- small utility kernels ----------------
__global__ void k_cvt_bf16_pad(const float* src, u16* dst, int rows_src, int cols, int rows_dst) {
  int i = blockIdx.x * blockDim.x + threadIdx.x;
  int n = rows_dst * cols;
  if (i >= n) return;
  int r = i / cols;
  dst[i] = (r < rows_src) ? f2bf(src[i]) : (u16)0;
}

__global__ void k_add_bias2(const float* a, const float* b, float* o, int n) {
  int i = blockIdx.x * blockDim.x + threadIdx.x;
  if (i < n) o[i] = a[i] + b[i];
}

__global__ void k_zero_u32(unsigned* p, int n) {
  for (int i = blockIdx.x * blockDim.x + threadIdx.x; i < n; i += gridDim.x * blockDim.x)
    p[i] = 0u;
}

// ---------------- LayerNorm (row = 768), f32 in -> bf16 out ----------------
__global__ void k_ln_f32(const float* __restrict__ X, u16* __restrict__ Y) {
  __shared__ float s1[256], s2[256];
  int row = blockIdx.x;
  int tid = threadIdx.x;
  const float* xr = X + (size_t)row * DD;
  float v0 = xr[tid], v1 = xr[tid + 256], v2 = xr[tid + 512];
  s1[tid] = v0 + v1 + v2;
  s2[tid] = v0 * v0 + v1 * v1 + v2 * v2;
  __syncthreads();
  for (int off = 128; off; off >>= 1) {
    if (tid < off) { s1[tid] += s1[tid + off]; s2[tid] += s2[tid + off]; }
    __syncthreads();
  }
  float mu = s1[0] * (1.f / DD);
  float var = s2[0] * (1.f / DD) - mu * mu;
  float rs = rsqrtf(var + 1e-5f);
  u16* yr = Y + (size_t)row * DD;
  yr[tid]       = f2bf((v0 - mu) * rs);
  yr[tid + 256] = f2bf((v1 - mu) * rs);
  yr[tid + 512] = f2bf((v2 - mu) * rs);
}

// ---------------- LayerNorm (row = 768), bf16 in -> bf16 out ----------------
__global__ void k_ln_bf16(const u16* __restrict__ X, u16* __restrict__ Y) {
  __shared__ float s1[256], s2[256];
  int row = blockIdx.x;
  int tid = threadIdx.x;
  const u16* xr = X + (size_t)row * DD;
  float v0 = bf2f(xr[tid]), v1 = bf2f(xr[tid + 256]), v2 = bf2f(xr[tid + 512]);
  s1[tid] = v0 + v1 + v2;
  s2[tid] = v0 * v0 + v1 * v1 + v2 * v2;
  __syncthreads();
  for (int off = 128; off; off >>= 1) {
    if (tid < off) { s1[tid] += s1[tid + off]; s2[tid] += s2[tid + off]; }
    __syncthreads();
  }
  float mu = s1[0] * (1.f / DD);
  float var = s2[0] * (1.f / DD) - mu * mu;
  float rs = rsqrtf(var + 1e-5f);
  u16* yr = Y + (size_t)row * DD;
  yr[tid]       = f2bf((v0 - mu) * rs);
  yr[tid + 256] = f2bf((v1 - mu) * rs);
  yr[tid + 512] = f2bf((v2 - mu) * rs);
}

// ---------------- input-projection GEMM: xg = Xln(32768x768) * W_ih^T + bias ----------------
// grid (24, 256, 2), block 256 (8 waves). Wave: 16 rows x 64 cols; K=768.
// sched_barrier(0) between the 10-load group and the 4-WMMA burst: all loads
// of a k-step issue in one clause (one s_wait), next k-step's loads are free
// to interleave with this k-step's WMMAs (software pipelining).
__global__ void k_xg_gemm(const u16* __restrict__ Xln,
                          const u16* __restrict__ Wf, const u16* __restrict__ Wb,
                          const float* __restrict__ biasf, const float* __restrict__ biasb,
                          u16* __restrict__ xgf, u16* __restrict__ xgb) {
  int dir = blockIdx.z;
  const u16* W = dir ? Wb : Wf;
  const float* bias = dir ? biasb : biasf;
  u16* xg = dir ? xgb : xgf;
  int lane = threadIdx.x & 31;
  int wave = threadIdx.x >> 5;
  int half = lane >> 4;
  int nl = lane & 15;
  int m0 = (blockIdx.y * 8 + wave) * 16;
  int n0 = blockIdx.x * 64;
  int mrow = m0 + nl;
  const u16* w0 = W + (n0 + nl) * DD;
  const u16* w1 = W + (n0 + 16 + nl) * DD;
  const u16* w2 = W + (n0 + 32 + nl) * DD;
  const u16* w3 = W + (n0 + 48 + nl) * DD;
  v8f acc0 = {}, acc1 = {}, acc2 = {}, acc3 = {};
#pragma unroll 4
  for (int k = 0; k < DD; k += 32) {
    v16bf a  = loadA(Xln, mrow, DD, k, lane);
    v16bf b0 = loadB(w0, k, lane);
    v16bf b1 = loadB(w1, k, lane);
    v16bf b2 = loadB(w2, k, lane);
    v16bf b3 = loadB(w3, k, lane);
    __builtin_prefetch(Xln + mrow * DD + k + 64, 0, 3);
    __builtin_amdgcn_sched_barrier(0);
    acc0 = wmma_bf16(a, b0, acc0);
    acc1 = wmma_bf16(a, b1, acc1);
    acc2 = wmma_bf16(a, b2, acc2);
    acc3 = wmma_bf16(a, b3, acc3);
  }
  v8f accs[4] = {acc0, acc1, acc2, acc3};
#pragma unroll
  for (int nt = 0; nt < 4; ++nt) {
    int n = n0 + nt * 16 + nl;
    float bv = bias[n];
#pragma unroll
    for (int r = 0; r < 8; ++r) {
      int m = m0 + half * 8 + r;
      xg[(size_t)m * G4H + n] = f2bf(accs[nt][r] + bv);
    }
  }
}

// ---------------- BiLSTM recurrence ----------------
// grid 24 blocks x 256 threads = 192 waves = 2 dirs x (4 mtiles x 24 jtiles).
// Each wave owns one 16x16 (batch x hidden) slice; all 4 gates in registers;
// cell state c stays in registers across all 512 steps. h double-buffered in
// global (L2-resident), device-wide sense-reversing barrier per timestep.
__global__ void k_lstm(const u16* __restrict__ xgf, const u16* __restrict__ xgb,
                       const u16* __restrict__ whhf, const u16* __restrict__ whhb,
                       u16* __restrict__ hbuf,   // [dir][parity][64][384] bf16
                       u16* __restrict__ hcat,   // [b*S+t][768] bf16
                       int* bar) {
  int lane = threadIdx.x & 31;
  int wave = threadIdx.x >> 5;
  int gw = blockIdx.x * 8 + wave;          // 0..191
  int dir = (gw >= 96) ? 1 : 0;
  int task = gw - dir * 96;                // 0..95
  int mtile = task & 3;                    // batch tile
  int jtile = task >> 2;                   // hidden tile (0..23)
  const u16* xg = dir ? xgb : xgf;
  const u16* Whh = dir ? whhb : whhf;
  u16* hb = hbuf + (size_t)dir * 2 * BB * HH;
  int half = lane >> 4;
  int nl = lane & 15;
  int mrow_a = mtile * 16 + nl;            // row for A loads
  int j = jtile * 16 + nl;                 // hidden column owned by lane
  int mbase = mtile * 16 + half * 8;       // rows for C/store
  const u16* wg0 = Whh + (0 * HH + j) * HH;
  const u16* wg1 = Whh + (1 * HH + j) * HH;
  const u16* wg2 = Whh + (2 * HH + j) * HH;
  const u16* wg3 = Whh + (3 * HH + j) * HH;
  v8f creg = {};
  int sense = 0;
  int nblk = (int)gridDim.x;

  for (int ts = 0; ts < SS; ++ts) {
    int t = dir ? (SS - 1 - ts) : ts;
    const u16* hsrc = hb + (ts & 1) * BB * HH;
    u16* hdst = hb + ((ts + 1) & 1) * BB * HH;
    v8f acc0 = {}, acc1 = {}, acc2 = {}, acc3 = {};
#pragma unroll 3
    for (int k = 0; k < HH; k += 32) {
      v16bf a  = loadA(hsrc, mrow_a, HH, k, lane);
      v16bf b0 = loadB(wg0, k, lane);
      v16bf b1 = loadB(wg1, k, lane);
      v16bf b2 = loadB(wg2, k, lane);
      v16bf b3 = loadB(wg3, k, lane);
      __builtin_amdgcn_sched_barrier(0);
      acc0 = wmma_bf16(a, b0, acc0);
      acc1 = wmma_bf16(a, b1, acc1);
      acc2 = wmma_bf16(a, b2, acc2);
      acc3 = wmma_bf16(a, b3, acc3);
    }
#pragma unroll
    for (int r = 0; r < 8; ++r) {
      int m = mbase + r;
      const u16* xrow = xg + (size_t)(m * SS + t) * G4H;
      float gi = acc0[r] + bf2f(xrow[0 * HH + j]);
      float gf = acc1[r] + bf2f(xrow[1 * HH + j]);
      float gg = acc2[r] + bf2f(xrow[2 * HH + j]);
      float go = acc3[r] + bf2f(xrow[3 * HH + j]);
      float c = sigf(gf) * creg[r] + sigf(gi) * tanhf(gg);
      float h = sigf(go) * tanhf(c);
      creg[r] = c;
      u16 hb16 = f2bf(h);
      hdst[m * HH + j] = hb16;
      hcat[(size_t)(m * SS + t) * DD + dir * HH + j] = hb16;
    }
    // device-wide barrier (24 resident blocks)
    __syncthreads();
    sense ^= 1;
    if (threadIdx.x == 0) {
      __threadfence();
      if (atomicAdd(bar, 1) == nblk - 1) {
        atomicExch(bar, 0);
        __threadfence();
        atomicExch(bar + 1, sense);
      } else {
        while (((volatile int*)bar)[1] != sense) __builtin_amdgcn_s_sleep(1);
      }
      __threadfence();
    }
    __syncthreads();
  }
}

// ---------------- FC1 (768->256) + LayerNorm(affine) + ReLU, fused ----------------
// grid 2048, block 256 (8 waves). Block: 16 rows x 256 cols; wave does ntiles {w, w+8}.
__global__ void k_fc1_ln_relu(const u16* __restrict__ X2, const u16* __restrict__ W1b,
                              const float* __restrict__ b1, const float* __restrict__ ln_g,
                              const float* __restrict__ ln_b, u16* __restrict__ h1) {
  __shared__ float tile[16 * 256];
  __shared__ float ps[256], pq[256];
  __shared__ float stats[32];
  int lane = threadIdx.x & 31;
  int wave = threadIdx.x >> 5;
  int half = lane >> 4;
  int nl = lane & 15;
  int m0 = blockIdx.x * 16;
  int mrow = m0 + nl;
  v8f acc0 = {}, acc1 = {};
  int nA = wave * 16 + nl;
  int nB = (wave + 8) * 16 + nl;
  const u16* wA = W1b + nA * DD;
  const u16* wB = W1b + nB * DD;
#pragma unroll 4
  for (int k = 0; k < DD; k += 32) {
    v16bf a  = loadA(X2, mrow, DD, k, lane);
    v16bf bA = loadB(wA, k, lane);
    v16bf bB = loadB(wB, k, lane);
    __builtin_amdgcn_sched_barrier(0);
    acc0 = wmma_bf16(a, bA, acc0);
    acc1 = wmma_bf16(a, bB, acc1);
  }
#pragma unroll
  for (int r = 0; r < 8; ++r) {
    int ml = half * 8 + r;
    tile[ml * 256 + nA] = acc0[r] + b1[nA];
    tile[ml * 256 + nB] = acc1[r] + b1[nB];
  }
  __syncthreads();
  int row = threadIdx.x >> 4;   // 0..15
  int seg = threadIdx.x & 15;   // 16-col segment
  float s = 0.f, q = 0.f;
#pragma unroll
  for (int i = 0; i < 16; ++i) {
    float v = tile[row * 256 + seg * 16 + i];
    s += v; q += v * v;
  }
  ps[threadIdx.x] = s; pq[threadIdx.x] = q;
  __syncthreads();
  if (seg == 0) {
    float S = 0.f, Q = 0.f;
#pragma unroll
    for (int i = 0; i < 16; ++i) { S += ps[row * 16 + i]; Q += pq[row * 16 + i]; }
    float mu = S * (1.f / 256.f);
    float var = Q * (1.f / 256.f) - mu * mu;
    stats[row * 2] = mu;
    stats[row * 2 + 1] = rsqrtf(var + 1e-5f);
  }
  __syncthreads();
  float mu = stats[row * 2], rs = stats[row * 2 + 1];
#pragma unroll
  for (int i = 0; i < 16; ++i) {
    int c = seg * 16 + i;
    float v = (tile[row * 256 + c] - mu) * rs * ln_g[c] + ln_b[c];
    h1[(size_t)(m0 + row) * 256 + c] = f2bf(fmaxf(v, 0.f));
  }
}

// ---------------- FC2 (256->128) + ReLU ----------------
// grid 2048, block 256 (8 waves): block = 16 rows x 128 cols, wave = one ntile.
__global__ void k_fc2_relu(const u16* __restrict__ h1, const u16* __restrict__ W2b,
                           const float* __restrict__ b2, u16* __restrict__ h2) {
  int lane = threadIdx.x & 31;
  int wave = threadIdx.x >> 5;
  int half = lane >> 4;
  int nl = lane & 15;
  int m0 = blockIdx.x * 16;
  int mrow = m0 + nl;
  int n = wave * 16 + nl;
  const u16* wn = W2b + n * 256;
  v8f acc = {};
#pragma unroll
  for (int k = 0; k < 256; k += 32) {
    v16bf a = loadA(h1, mrow, 256, k, lane);
    v16bf b = loadB(wn, k, lane);
    __builtin_amdgcn_sched_barrier(0);
    acc = wmma_bf16(a, b, acc);
  }
  float bv = b2[n];
#pragma unroll
  for (int r = 0; r < 8; ++r) {
    int m = m0 + half * 8 + r;
    h2[(size_t)m * 128 + n] = f2bf(fmaxf(acc[r] + bv, 0.f));
  }
}

// ---------------- FC3 (128->3): W3 zero-padded to 16 rows so EXEC stays all-ones ----------------
// grid 256, block 256 (8 waves): wave = one 16-row tile.
__global__ void k_fc3(const u16* __restrict__ h2, const u16* __restrict__ W3b,
                      const float* __restrict__ b3, float* __restrict__ O) {
  int lane = threadIdx.x & 31;
  int wave = threadIdx.x >> 5;
  int half = lane >> 4;
  int nl = lane & 15;
  int m0 = (blockIdx.x * 8 + wave) * 16;
  int mrow = m0 + nl;
  const u16* wn = W3b + nl * 128;
  v8f acc = {};
#pragma unroll
  for (int k = 0; k < 128; k += 32) {
    v16bf a = loadA(h2, mrow, 128, k, lane);
    v16bf b = loadB(wn, k, lane);
    __builtin_amdgcn_sched_barrier(0);
    acc = wmma_bf16(a, b, acc);
  }
  if (nl < CC) {
    float bv = b3[nl];
#pragma unroll
    for (int r = 0; r < 8; ++r) {
      int m = m0 + half * 8 + r;
      O[(size_t)m * CC + nl] = acc[r] + bv;
    }
  }
}

// ---------------- CRF log-likelihood (tiny): 1 block, 1 thread per batch row ----------------
__global__ void k_crf(const float* __restrict__ O, const int* __restrict__ labels,
                      const float* __restrict__ start, const float* __restrict__ endt,
                      const float* __restrict__ trans, float* __restrict__ llh) {
  __shared__ float part[BB];
  int b = threadIdx.x;
  const float* Ob = O + (size_t)b * SS * CC;
  const int* lb = labels + (size_t)b * SS;
  int t0 = lb[0]; if (t0 < 0) t0 = 0;
  float num = start[t0] + Ob[t0];
  int prev = t0, last = t0;
  float sc0 = start[0] + Ob[0];
  float sc1 = start[1] + Ob[1];
  float sc2 = start[2] + Ob[2];
  for (int t = 1; t < SS; ++t) {
    int tag = lb[t];
    if (tag >= 0) {
      const float* emt = Ob + t * CC;
      num += trans[prev * CC + tag] + emt[tag];
      prev = tag; last = tag;
      float n0 = lse3(sc0 + trans[0], sc1 + trans[CC], sc2 + trans[2 * CC]) + emt[0];
      float n1 = lse3(sc0 + trans[1], sc1 + trans[CC + 1], sc2 + trans[2 * CC + 1]) + emt[1];
      float n2 = lse3(sc0 + trans[2], sc1 + trans[CC + 2], sc2 + trans[2 * CC + 2]) + emt[2];
      sc0 = n0; sc1 = n1; sc2 = n2;
    }
  }
  num += endt[last];
  float den = lse3(sc0 + endt[0], sc1 + endt[1], sc2 + endt[2]);
  part[b] = num - den;
  __syncthreads();
  if (b == 0) {
    float s = 0.f;
    for (int i = 0; i < BB; ++i) s += part[i];
    llh[0] = s;
  }
}

// ---------------- host launcher ----------------
extern "C" void kernel_launch(void* const* d_in, const int* in_sizes, int n_in,
                              void* d_out, int out_size, void* d_ws, size_t ws_size,
                              hipStream_t stream) {
  (void)in_sizes; (void)n_in; (void)out_size; (void)ws_size;
  const float* emb    = (const float*)d_in[0];
  const int*   labels = (const int*)d_in[1];
  const float* w_ih_f = (const float*)d_in[2];
  const float* w_hh_f = (const float*)d_in[3];
  const float* b_ih_f = (const float*)d_in[4];
  const float* b_hh_f = (const float*)d_in[5];
  const float* w_ih_b = (const float*)d_in[6];
  const float* w_hh_b = (const float*)d_in[7];
  const float* b_ih_b = (const float*)d_in[8];
  const float* b_hh_b = (const float*)d_in[9];
  const float* W1     = (const float*)d_in[10];
  const float* b1     = (const float*)d_in[11];
  const float* ln_g   = (const float*)d_in[12];
  const float* ln_b   = (const float*)d_in[13];
  const float* W2     = (const float*)d_in[14];
  const float* b2     = (const float*)d_in[15];
  const float* W3     = (const float*)d_in[16];
  const float* b3     = (const float*)d_in[17];
  const float* startt = (const float*)d_in[18];
  const float* endt   = (const float*)d_in[19];
  const float* trans  = (const float*)d_in[20];

  float* out = (float*)d_out;        // out[0] = llh, out[1..] = O (32768 x 3)
  float* O = out + 1;

  // workspace carve-out
  char* w = (char*)d_ws;
  size_t off = 0;
  auto take = [&](size_t bytes) -> char* {
    char* p = w + off;
    off = (off + bytes + 255) & ~(size_t)255;
    return p;
  };
  u16* Xln   = (u16*)take((size_t)BS * DD * 2);
  u16* wihf  = (u16*)take((size_t)G4H * DD * 2);
  u16* wihb  = (u16*)take((size_t)G4H * DD * 2);
  u16* whhf  = (u16*)take((size_t)G4H * HH * 2);
  u16* whhb  = (u16*)take((size_t)G4H * HH * 2);
  u16* W1b   = (u16*)take((size_t)256 * DD * 2);
  u16* W2b   = (u16*)take((size_t)128 * 256 * 2);
  u16* W3b   = (u16*)take((size_t)16 * 128 * 2);
  float* biasf = (float*)take((size_t)G4H * 4);
  float* biasb = (float*)take((size_t)G4H * 4);
  u16* xgf   = (u16*)take((size_t)BS * G4H * 2);
  u16* xgb   = (u16*)take((size_t)BS * G4H * 2);
  u16* hbuf  = (u16*)take((size_t)2 * 2 * BB * HH * 2);
  int* bar   = (int*)take(256);
  u16* hcat  = (u16*)take((size_t)BS * DD * 2);
  u16* X2    = (u16*)take((size_t)BS * DD * 2);
  u16* h1    = (u16*)take((size_t)BS * 256 * 2);
  u16* h2    = (u16*)take((size_t)BS * 128 * 2);

  // 1. weight conversions to bf16 (+ W3 zero-pad to 16 rows)
  auto cvt = [&](const float* s, u16* d, int rs, int cols, int rd) {
    int n = rd * cols;
    k_cvt_bf16_pad<<<(n + 255) / 256, 256, 0, stream>>>(s, d, rs, cols, rd);
  };
  cvt(w_ih_f, wihf, G4H, DD, G4H);
  cvt(w_ih_b, wihb, G4H, DD, G4H);
  cvt(w_hh_f, whhf, G4H, HH, G4H);
  cvt(w_hh_b, whhb, G4H, HH, G4H);
  cvt(W1, W1b, 256, DD, 256);
  cvt(W2, W2b, 128, 256, 128);
  cvt(W3, W3b, CC, 128, 16);
  k_add_bias2<<<(G4H + 255) / 256, 256, 0, stream>>>(b_ih_f, b_hh_f, biasf, G4H);
  k_add_bias2<<<(G4H + 255) / 256, 256, 0, stream>>>(b_ih_b, b_hh_b, biasb, G4H);

  // 2. LayerNorm of embeddings -> bf16
  k_ln_f32<<<BS, 256, 0, stream>>>(emb, Xln);

  // 3. input-projection GEMMs (both directions), fully parallel
  k_xg_gemm<<<dim3(G4H / 64, BS / 128, 2), 256, 0, stream>>>(Xln, wihf, wihb, biasf, biasb, xgf, xgb);

  // 4. zero h double-buffers and barrier state
  k_zero_u32<<<64, 256, 0, stream>>>((unsigned*)hbuf, (2 * 2 * BB * HH * 2) / 4);
  k_zero_u32<<<1, 32, 0, stream>>>((unsigned*)bar, 2);

  // 5. BiLSTM recurrence: 24 persistent blocks, device barrier per timestep
  k_lstm<<<24, 256, 0, stream>>>(xgf, xgb, whhf, whhb, hbuf, hcat, bar);

  // 6. LayerNorm of concat(hf,hb)
  k_ln_bf16<<<BS, 256, 0, stream>>>(hcat, X2);

  // 7-9. MLP head
  k_fc1_ln_relu<<<BS / 16, 256, 0, stream>>>(X2, W1b, b1, ln_g, ln_b, h1);
  k_fc2_relu<<<BS / 16, 256, 0, stream>>>(h1, W2b, b2, h2);
  k_fc3<<<BS / 128, 256, 0, stream>>>(h2, W3b, b3, O);

  // 10. CRF log-likelihood
  k_crf<<<1, BB, 0, stream>>>(O, labels, startt, endt, trans, out);
}
